// EncoderRNN_46772193853590
// MI455X (gfx1250) — matrix operand order repaired and checked by reading, not compiled, not run on hardware
//
#include <hip/hip_runtime.h>
#include <math.h>

// ---------------- problem dims ----------------
#define VOCAB   32000
#define HIDDEN  1024
#define SEQLEN  4096
#define H3      (3 * HIDDEN)            // 3072

// ---------------- workspace layout (bytes) ----------------
#define OFF_GI   0ull                                    // 4096*3072*4  = 50331648
#define OFF_XS   (OFF_GI  + (size_t)SEQLEN * H3 * 4)     // + 8 MB xs bf16
#define OFF_WIH  (OFF_XS  + (size_t)SEQLEN * HIDDEN * 2) // + 6 MB w_ih bf16
#define OFF_H    (OFF_WIH + (size_t)H3 * HIDDEN * 2)     // + 8 KB h double buffer
#define OFF_CNT  (OFF_H   + (size_t)2 * HIDDEN * 4)      // + 16 KB counters

// ---------------- types ----------------
typedef __bf16 bf16_t;
typedef __attribute__((ext_vector_type(8)))  __bf16 v8bf;
typedef __attribute__((ext_vector_type(16))) __bf16 v16bf;
typedef __attribute__((ext_vector_type(8)))  float  v8f;

// ---------------- init: zero h buffers + barrier counters ----------------
__global__ void zero_init(unsigned* p, int n) {
    int i = blockIdx.x * blockDim.x + threadIdx.x;
    if (i < n) p[i] = 0u;
}

// ---------------- embedding gather + fp32 -> bf16 ----------------
__global__ __launch_bounds__(256) void gather_cast(const float* __restrict__ emb,
                                                   const int* __restrict__ x,
                                                   bf16_t* __restrict__ xs) {
    int t   = blockIdx.x;
    int tok = x[t];
    const float* src = emb + (size_t)tok * HIDDEN;
    bf16_t* dst = xs + (size_t)t * HIDDEN;
    for (int i = threadIdx.x; i < HIDDEN; i += 256)
        dst[i] = (bf16_t)src[i];
}

// ---------------- generic fp32 -> bf16 cast (for w_ih) ----------------
__global__ __launch_bounds__(256) void cast_f32_bf16(const float* __restrict__ src,
                                                     bf16_t* __restrict__ dst, int n) {
    int stride = gridDim.x * blockDim.x;
    for (int i = blockIdx.x * blockDim.x + threadIdx.x; i < n; i += stride)
        dst[i] = (bf16_t)src[i];
}

// ---------------- WMMA helper ----------------
static __device__ inline v8f wmma_bf16(v16bf a, v16bf b, v8f c) {
    return __builtin_amdgcn_wmma_f32_16x16x32_bf16(
        /*neg_a=*/false, a, /*neg_b=*/false, b,
        /*c_mod=*/(short)0, c, /*reuse_a=*/false, /*reuse_b=*/false);
}

// Build v16bf A-fragment per ISA 16-bit A(16x32) layout:
//  lanes 0-15:  K = kk+0..7   and kk+16..23   (M = lane)
//  lanes 16-31: K = kk+8..15  and kk+24..31   (M = lane-16)
static __device__ inline v16bf load_a_frag(const bf16_t* row, int kk, int hi) {
    int ka = kk + (hi ? 8 : 0);
    v8bf lo = *(const v8bf*)(row + ka);
    v8bf hi8 = *(const v8bf*)(row + ka + 16);
    v16bf a;
#pragma unroll
    for (int i = 0; i < 8; ++i) { a[i] = lo[i]; a[i + 8] = hi8[i]; }
    return a;
}

// Build v16bf B-fragment per ISA 16-bit B(32x16) layout:
//  lanes 0-15:  K = kk+0..7 and kk+16..23 (N = lane);  lanes 16-31: K = kk+8..15 and kk+24..31 (N = lane-16)
//  B[k][n] = w_ih[n][k]  -> lane reads two 8-bf16 chunks from w_ih row n.
static __device__ inline v16bf load_b_frag(const bf16_t* row, int kk, int hi) {
    int kb = kk + (hi ? 8 : 0);
    v8bf b0 = *(const v8bf*)(row + kb);
    v8bf b1 = *(const v8bf*)(row + kb + 16);
    v16bf b;
#pragma unroll
    for (int i = 0; i < 8; ++i) { b[i] = b0[i]; b[i + 8] = b1[i]; }
    return b;
}

// ---------------- gi_all = xs @ w_ih^T + b_ih  via WMMA bf16 ----------------
// grid: (4096/16, 3072/256), block: 128 (4 waves). Wave w computes a 16x64 strip.
__global__ __launch_bounds__(128) void wmma_gemm(const bf16_t* __restrict__ xs,
                                                 const bf16_t* __restrict__ wih,
                                                 const float*  __restrict__ bih,
                                                 float* __restrict__ gi) {
    const int wave = threadIdx.x >> 5;
    const int lane = threadIdx.x & 31;
    const int lr   = lane & 15;
    const int hi   = lane >> 4;

    const int m     = blockIdx.x * 16;
    const int nbase = blockIdx.y * 256 + wave * 64;

    const bf16_t* arow = xs + (size_t)(m + lr) * HIDDEN;
    const bf16_t* brow[4];
    v8f c[4];
#pragma unroll
    for (int s = 0; s < 4; ++s) {
        int n = nbase + s * 16 + lr;
        brow[s] = wih + (size_t)n * HIDDEN;
        float bias = bih[n];
#pragma unroll
        for (int j = 0; j < 8; ++j) c[s][j] = bias;   // bias broadcast down column
    }

    for (int kk = 0; kk < HIDDEN; kk += 32) {
        v16bf a = load_a_frag(arow, kk, hi);
#pragma unroll
        for (int s = 0; s < 4; ++s) {
            v16bf b = load_b_frag(brow[s], kk, hi);
            c[s] = wmma_bf16(a, b, c[s]);
        }
    }

    // C/D layout: VGPR j, lanes 0-15 -> (M=j, N=lane); lanes 16-31 -> (M=8+j, N=lane-16)
#pragma unroll
    for (int s = 0; s < 4; ++s) {
        int n = nbase + s * 16 + lr;
#pragma unroll
        for (int j = 0; j < 8; ++j) {
            int row = m + j + 8 * hi;
            gi[(size_t)row * H3 + n] = c[s][j];
        }
    }
}

// ---------------- sequential GRU scan ----------------
// 128 persistent workgroups x 256 threads. WG b owns h[b*8 .. b*8+8) and the 24
// matching rows of w_hh (r/z/n gates), pinned in LDS in fp32. One wave per
// hidden element: 32-lane K-split dot products + shfl_xor reduction.
#define NWG   128
#define HPER  8            // 1024 / 128
#define ROWS  (3 * HPER)   // 24

__global__ __launch_bounds__(256) void gru_scan(const float* __restrict__ w_hh,
                                                const float* __restrict__ b_hh,
                                                const float* __restrict__ gi,
                                                float* hbuf, unsigned* counters,
                                                float* __restrict__ out) {
    extern __shared__ float smem[];
    float* wlds = smem;                 // ROWS * 1024 fp32 = 96 KB
    float* hlds = smem + ROWS * HIDDEN; // 1024 fp32

    const int tid   = threadIdx.x;
    const int hbase = blockIdx.x * HPER;

    // Pin this WG's w_hh rows in LDS (fp32, read once from HBM/L2).
    for (int i = tid; i < ROWS * HIDDEN; i += 256) {
        int local = i >> 10;            // 0..23
        int k     = i & (HIDDEN - 1);
        int gate  = local / HPER;       // 0:r 1:z 2:n
        int e     = local % HPER;
        wlds[i] = w_hh[(size_t)(gate * HIDDEN + hbase + e) * HIDDEN + k];
    }
    __syncthreads();

    const int wave = tid >> 5;          // 0..7  -> local h element
    const int lane = tid & 31;
    const int e    = wave;
    const int ge   = hbase + e;         // global h index

    const float b_r = b_hh[ge];
    const float b_z = b_hh[HIDDEN + ge];
    const float b_n = b_hh[2 * HIDDEN + ge];

    const float* wr = &wlds[(0 * HPER + e) * HIDDEN];
    const float* wz = &wlds[(1 * HPER + e) * HIDDEN];
    const float* wn = &wlds[(2 * HPER + e) * HIDDEN];

    float h_keep = 0.0f;

    for (int t = 0; t < SEQLEN; ++t) {
        const float* hcur = hbuf + (t & 1) * HIDDEN;
        float*       hnxt = hbuf + ((t + 1) & 1) * HIDDEN;

        // Stage full h into LDS with agent-scope loads (never hit stale WGP$).
        for (int i = tid; i < HIDDEN; i += 256)
            hlds[i] = __hip_atomic_load(hcur + i, __ATOMIC_RELAXED,
                                        __HIP_MEMORY_SCOPE_AGENT);
        __syncthreads();

        float ar = 0.f, az = 0.f, an = 0.f;
        for (int k = lane; k < HIDDEN; k += 32) {
            float hv = hlds[k];
            ar = fmaf(wr[k], hv, ar);
            az = fmaf(wz[k], hv, az);
            an = fmaf(wn[k], hv, an);
        }
#pragma unroll
        for (int m = 16; m; m >>= 1) {
            ar += __shfl_xor(ar, m, 32);
            az += __shfl_xor(az, m, 32);
            an += __shfl_xor(an, m, 32);
        }

        if (lane == 0) {
            const float* git = gi + (size_t)t * H3;
            float i_r = git[ge];
            float i_z = git[HIDDEN + ge];
            float i_n = git[2 * HIDDEN + ge];
            float r = 1.0f / (1.0f + expf(-(i_r + ar + b_r)));
            float z = 1.0f / (1.0f + expf(-(i_z + az + b_z)));
            float n = tanhf(i_n + r * (an + b_n));
            float hn = (1.0f - z) * n + z * hlds[ge];
            h_keep = hn;
            __hip_atomic_store(hnxt + ge, hn, __ATOMIC_RELEASE,
                               __HIP_MEMORY_SCOPE_AGENT);
        }
        __syncthreads();

        // Grid-wide step barrier: per-step arrival counter, acquire-spin.
        if (tid == 0) {
            __hip_atomic_fetch_add(&counters[t], 1u, __ATOMIC_ACQ_REL,
                                   __HIP_MEMORY_SCOPE_AGENT);
            while (__hip_atomic_load(&counters[t], __ATOMIC_ACQUIRE,
                                     __HIP_MEMORY_SCOPE_AGENT) < (unsigned)NWG)
                __builtin_amdgcn_s_sleep(1);
        }
        __syncthreads();
    }

    if (lane == 0) out[ge] = h_keep;
}

// ---------------- host launch ----------------
extern "C" void kernel_launch(void* const* d_in, const int* in_sizes, int n_in,
                              void* d_out, int out_size, void* d_ws, size_t ws_size,
                              hipStream_t stream) {
    const float* emb  = (const float*)d_in[0];
    const float* w_ih = (const float*)d_in[1];
    const float* w_hh = (const float*)d_in[2];
    const float* b_ih = (const float*)d_in[3];
    const float* b_hh = (const float*)d_in[4];
    const int*   x    = (const int*)d_in[5];
    float*       out  = (float*)d_out;

    char* ws = (char*)d_ws;
    float*    gi       = (float*)(ws + OFF_GI);
    bf16_t*   xs       = (bf16_t*)(ws + OFF_XS);
    bf16_t*   wihb     = (bf16_t*)(ws + OFF_WIH);
    float*    hbuf     = (float*)(ws + OFF_H);
    unsigned* counters = (unsigned*)(ws + OFF_CNT);

    // h buffers (2*1024 fp32) + counters (4096 u32) are contiguous: 6144 words.
    zero_init<<<24, 256, 0, stream>>>((unsigned*)(ws + OFF_H), 6144);

    gather_cast<<<SEQLEN, 256, 0, stream>>>(emb, x, xs);
    cast_f32_bf16<<<768, 256, 0, stream>>>(w_ih, wihb, H3 * HIDDEN);

    wmma_gemm<<<dim3(SEQLEN / 16, H3 / 256), 128, 0, stream>>>(xs, wihb, b_ih, gi);

    size_t lds_bytes = (size_t)(ROWS * HIDDEN + HIDDEN) * sizeof(float); // 100 KB
    gru_scan<<<NWG, 256, lds_bytes, stream>>>(w_hh, b_hh, gi, hbuf, counters, out);
}
